// UDAGCN_42734924595737
// MI455X (gfx1250) — matrix operand order repaired and checked by reading, not compiled, not run on hardware
//
#include <hip/hip_runtime.h>
#include <hip/hip_bf16.h>

// ---------------------------------------------------------------------------
// UDAGCN forward for MI455X (gfx1250, wave32, WMMA).
//   GEMMs: v_wmma_f32_16x16x32_bf16 (f32 accumulate). One wave computes a full
//          16 x NOUT row stripe (NOUT/16 accumulators), so X streams from HBM
//          exactly once; packed-B fragments (<=128KB) stay cache-resident and
//          are fetched as coalesced global_load_b128.
//   Aggregation: one wave32 per edge, coalesced lane-strided features,
//                global_atomic_add_f32 scatter (L2-resident working set).
// ---------------------------------------------------------------------------

typedef __bf16 bf16_t;
typedef __bf16 v16bf __attribute__((ext_vector_type(16)));
typedef float  v8f   __attribute__((ext_vector_type(8)));

#define F_IN 512
#define HID  128
#define ENC  64
#define NCLS 10
#define DOMH 40

// ---------------- weight packing: W[K,NOUT] f32 -> WMMA B-fragment bf16 ------
// Packed layout: for column-tile t (NOUT/16) and K-block kb (K/32):
//   out[((t*(K/32) + kb)*32 + lane)*16 + i] = W[(kb*32 + (lane>>4)*16 + i)*NOUT
//                                               + t*16 + (lane&15)]
// so a wave's B fragment for (t,kb) is one contiguous 1KB block, 32B per lane.

template <int K, int NOUT>
__global__ void pack_b_kernel(const float* __restrict__ W, bf16_t* __restrict__ out) {
    int idx = blockIdx.x * blockDim.x + threadIdx.x;
    if (idx >= K * NOUT) return;
    int i    = idx & 15;
    int lane = (idx >> 4) & 31;
    int kb   = (idx >> 9) % (K / 32);
    int t    = idx / (512 * (K / 32));
    int k    = kb * 32 + (lane >> 4) * 16 + i;
    int col  = t * 16 + (lane & 15);
    out[idx] = (bf16_t)W[(size_t)k * NOUT + col];
}

// degrees for both normalizations (includes implicit self loops, weight 1)
__global__ void degree_kernel(const long long* __restrict__ ei,   // [2,E] flat
                              const float* __restrict__ pw,       // ppmi weights [E]
                              float* __restrict__ degG,
                              float* __restrict__ degP, int E, int N) {
    int e = blockIdx.x * blockDim.x + threadIdx.x;
    if (e < E) {
        int d = (int)ei[(size_t)E + e];
        atomicAdd(&degG[d], 1.0f);
        atomicAdd(&degP[d], pw[e]);
    } else if (e < E + N) {
        int d = e - E;
        atomicAdd(&degG[d], 1.0f);
        atomicAdd(&degP[d], 1.0f);
    }
}

// deg -> d^{-1/2} in place, for both norms
__global__ void rsqrt_kernel(float* __restrict__ dg, float* __restrict__ dp, int N) {
    int i = blockIdx.x * blockDim.x + threadIdx.x;
    if (i >= N) return;
    float a = dg[i]; dg[i] = (a > 0.0f) ? rsqrtf(a) : 0.0f;
    float b = dp[i]; dp[i] = (b > 0.0f) ? rsqrtf(b) : 0.0f;
}

// ---------------- WMMA GEMM: Y[nrows,NOUT] = X[nrows,K] @ W[K,NOUT] ----------
// One wave computes a full 16 x NOUT row stripe: NOUT/16 v8f accumulators,
// A fragment loaded once per 32-deep K-block (X streamed exactly once),
// B fragments from the cache-resident packed stream (2x b128 per tile).

template <int K, int NOUT>
__global__ __launch_bounds__(32)
void gemm_wmma_bf16(const float* __restrict__ X, const bf16_t* __restrict__ Wpk,
                    float* __restrict__ Y, int nrows) {
    constexpr int NT = NOUT / 16;    // column tiles per wave
    constexpr int KB = K / 32;       // K blocks

    const int tileM = blockIdx.x;
    if (tileM * 16 >= nrows) return;

    const int lane = threadIdx.x;    // 0..31
    const int lo   = lane & 15;
    const int hi   = lane >> 4;      // 0 or 1

    v8f c[NT];
#pragma unroll
    for (int t = 0; t < NT; ++t) c[t] = {};

    // A: row = tileM*16 + lo ; this lane supplies K = hi*16+{0..15}
    const float* xr = X + (size_t)(tileM * 16 + lo) * K + hi * 16;

    for (int kb = 0; kb < KB; ++kb) {
        v16bf a;
#pragma unroll
        for (int i = 0; i < 16; ++i) {
            a[i]     = (bf16_t)xr[kb * 32 + i];
        }
#pragma unroll
        for (int t = 0; t < NT; ++t) {
            // contiguous 32B per lane -> two global_load_b128, fully coalesced
            v16bf b = *(const v16bf*)(Wpk + ((size_t)(t * KB + kb) * 32 + lane) * 16);
            // independent accumulator chains: no WMMA->WMMA RAW hazards
            c[t] = __builtin_amdgcn_wmma_f32_16x16x32_bf16(
                       false, a, false, b, (short)0, c[t], false, false);
        }
    }

    // C/D layout: VGPR v, lane -> row = v + 8*hi, col = t*16 + lo
    float* yr = Y + (size_t)(tileM * 16 + hi * 8) * NOUT + lo;
#pragma unroll
    for (int t = 0; t < NT; ++t)
#pragma unroll
        for (int v = 0; v < 8; ++v)
            yr[(size_t)v * NOUT + t * 16] = c[t][v];
}

// ---------------- edge aggregation: Aout[dst] += Hin[src] * enorm ------------
// One wave per edge (edges [0,E) real, [E,E+N) self loops). Lanes stride F.

template <int F>
__global__ void aggregate_kernel(const long long* __restrict__ ei,
                                 const float* __restrict__ ew,   // nullptr -> 1
                                 const float* __restrict__ dinv,
                                 const float* __restrict__ Hin,
                                 float* __restrict__ Aout, int E, int N) {
    long long gid = (long long)blockIdx.x * blockDim.x + threadIdx.x;
    int wave = (int)(gid >> 5);
    int lane = (int)(gid & 31);
    if (wave >= E + N) return;

    int s, d; float w;
    if (wave < E) {
        s = (int)ei[wave];
        d = (int)ei[(size_t)E + wave];
        float base = ew ? ew[wave] : 1.0f;
        w = base * dinv[s] * dinv[d];
    } else {
        s = d = wave - E;
        float di = dinv[s];
        w = di * di;
    }
    const float* hs = Hin + (size_t)s * F;
    float* ad = Aout + (size_t)d * F;
#pragma unroll
    for (int f = lane; f < F; f += 32)
        atomicAdd(&ad[f], hs[f] * w);
}

// bias add (+ optional relu) in place; F must be a power of two
__global__ void bias_act_kernel(float* __restrict__ H, const float* __restrict__ b,
                                long long total, int F, int relu) {
    long long i = (long long)blockIdx.x * blockDim.x + threadIdx.x;
    if (i >= total) return;
    float v = H[i] + b[(int)i & (F - 1)];
    if (relu) v = fmaxf(v, 0.0f);
    H[i] = v;
}

// ---------------- fused head: attention, classifier, domain -----------------

__global__ void head_kernel(const float* __restrict__ g, const float* __restrict__ p,
                            const float* __restrict__ att_w, const float* __restrict__ att_b,
                            const float* __restrict__ cls_w, const float* __restrict__ cls_b,
                            const float* __restrict__ dw1, const float* __restrict__ db1,
                            const float* __restrict__ dw2, const float* __restrict__ db2,
                            float* __restrict__ emb_out, float* __restrict__ cls_out,
                            float* __restrict__ dom_out, int N) {
    int i = blockIdx.x * blockDim.x + threadIdx.x;
    if (i >= N) return;
    const float* gi = g + (size_t)i * ENC;
    const float* pi = p + (size_t)i * ENC;

    // attention logits over the two branches
    float lg = att_b[0], lp = att_b[0];
#pragma unroll
    for (int k = 0; k < ENC; ++k) {
        float aw = att_w[k];
        lg += gi[k] * aw;
        lp += pi[k] * aw;
    }
    float m  = fmaxf(lg, lp);
    float eg = __expf(lg - m), ep = __expf(lp - m);
    float inv = 1.0f / (eg + ep);
    float wg = eg * inv, wp = ep * inv;

    float e[ENC];
#pragma unroll
    for (int k = 0; k < ENC; ++k) {
        e[k] = wg * gi[k] + wp * pi[k];
        emb_out[(size_t)i * ENC + k] = e[k];
    }

    // classifier head
#pragma unroll
    for (int c = 0; c < NCLS; ++c) {
        float acc = cls_b[c];
#pragma unroll
        for (int k = 0; k < ENC; ++k) acc += e[k] * cls_w[k * NCLS + c];
        cls_out[(size_t)i * NCLS + c] = acc;
    }

    // domain head (grad-reversal is identity in forward)
    float h[DOMH];
#pragma unroll
    for (int j = 0; j < DOMH; ++j) {
        float acc = db1[j];
#pragma unroll
        for (int k = 0; k < ENC; ++k) acc += e[k] * dw1[k * DOMH + j];
        h[j] = fmaxf(acc, 0.0f);
    }
#pragma unroll
    for (int c = 0; c < 2; ++c) {
        float acc = db2[c];
#pragma unroll
        for (int j = 0; j < DOMH; ++j) acc += h[j] * dw2[j * 2 + c];
        dom_out[(size_t)i * 2 + c] = acc;
    }
}

// ---------------------------------------------------------------------------

extern "C" void kernel_launch(void* const* d_in, const int* in_sizes, int n_in,
                              void* d_out, int out_size, void* d_ws, size_t ws_size,
                              hipStream_t stream) {
    const float*      x     = (const float*)d_in[0];
    const long long*  ei    = (const long long*)d_in[1];   // int64 [2,E]
    const float*      ppmi  = (const float*)d_in[2];
    /* d_in[3] = rate (unused: gradient reversal is identity forward) */
    const float* W1    = (const float*)d_in[4];
    const float* b1    = (const float*)d_in[5];
    const float* W2    = (const float*)d_in[6];
    const float* b2    = (const float*)d_in[7];
    const float* att_w = (const float*)d_in[8];
    const float* att_b = (const float*)d_in[9];
    const float* cls_w = (const float*)d_in[10];
    const float* cls_b = (const float*)d_in[11];
    const float* dw1   = (const float*)d_in[12];
    const float* db1   = (const float*)d_in[13];
    const float* dw2   = (const float*)d_in[14];
    const float* db2   = (const float*)d_in[15];

    const int N = in_sizes[0] / F_IN;
    const int E = in_sizes[1] / 2;

    // ---- workspace carve-out (with buffer reuse; ~155 MB total) ----
    char* ws = (char*)d_ws;
    size_t off = 0;
    auto carve = [&](size_t bytes) -> char* {
        char* pt = ws + off;
        off += (bytes + 255) & ~(size_t)255;
        return pt;
    };
    float*  degG = (float*)carve((size_t)N * 4);             // becomes dinv_g
    float*  degP = (float*)carve((size_t)N * 4);             // becomes dinv_p
    bf16_t* w1pk = (bf16_t*)carve((size_t)F_IN * HID * 2);   // packed B frags
    bf16_t* w2pk = (bf16_t*)carve((size_t)HID * ENC * 2);
    float*  bufA = (float*)carve((size_t)N * HID * 4);       // XW1, later XW2g|XW2p
    float*  bufB = (float*)carve((size_t)N * HID * 4);       // H1g, later g|p
    float*  bufC = (float*)carve((size_t)N * HID * 4);       // H1p

    const int threads = 256;
    const long long EW = (long long)(E + N);                 // edges incl. self loops

    // 1) degree accumulators -> d^{-1/2}
    hipMemsetAsync(degG, 0, (size_t)N * 4, stream);
    hipMemsetAsync(degP, 0, (size_t)N * 4, stream);
    degree_kernel<<<(unsigned)((EW + threads - 1) / threads), threads, 0, stream>>>(
        ei, ppmi, degG, degP, E, N);
    rsqrt_kernel<<<(N + threads - 1) / threads, threads, 0, stream>>>(degG, degP, N);

    // 2) weights -> bf16 WMMA B-fragment order, once
    pack_b_kernel<F_IN, HID><<<(F_IN * HID + threads - 1) / threads, threads, 0, stream>>>(
        W1, w1pk);
    pack_b_kernel<HID, ENC><<<(HID * ENC + threads - 1) / threads, threads, 0, stream>>>(
        W2, w2pk);

    // 3) XW1 = X @ W1  (WMMA bf16; one wave per 16-row stripe, X streamed once)
    gemm_wmma_bf16<F_IN, HID><<<(N + 15) / 16, 32, 0, stream>>>(x, w1pk, bufA, N);

    // 4) H1g = relu(agg_gcn(XW1) + b1), H1p = relu(agg_ppmi(XW1) + b1)
    const unsigned aggBlocks = (unsigned)((EW * 32 + threads - 1) / threads);
    hipMemsetAsync(bufB, 0, (size_t)N * HID * 4, stream);
    aggregate_kernel<HID><<<aggBlocks, threads, 0, stream>>>(ei, nullptr, degG, bufA, bufB, E, N);
    bias_act_kernel<<<(unsigned)(((long long)N * HID + threads - 1) / threads), threads, 0, stream>>>(
        bufB, b1, (long long)N * HID, HID, 1);

    hipMemsetAsync(bufC, 0, (size_t)N * HID * 4, stream);
    aggregate_kernel<HID><<<aggBlocks, threads, 0, stream>>>(ei, ppmi, degP, bufA, bufC, E, N);
    bias_act_kernel<<<(unsigned)(((long long)N * HID + threads - 1) / threads), threads, 0, stream>>>(
        bufC, b1, (long long)N * HID, HID, 1);

    // 5) XW2g = H1g @ W2 ; XW2p = H1p @ W2   (reuse bufA — XW1 fully consumed)
    float* XW2g = bufA;
    float* XW2p = bufA + (size_t)N * ENC;
    gemm_wmma_bf16<HID, ENC><<<(N + 15) / 16, 32, 0, stream>>>(bufB, w2pk, XW2g, N);
    gemm_wmma_bf16<HID, ENC><<<(N + 15) / 16, 32, 0, stream>>>(bufC, w2pk, XW2p, N);

    // 6) g = agg_gcn(XW2g)+b2 ; p = agg_ppmi(XW2p)+b2   (reuse bufB — H1g consumed)
    float* gbuf = bufB;
    float* pbuf = bufB + (size_t)N * ENC;
    hipMemsetAsync(bufB, 0, (size_t)N * HID * 4, stream);   // covers both g and p
    aggregate_kernel<ENC><<<aggBlocks, threads, 0, stream>>>(ei, nullptr, degG, XW2g, gbuf, E, N);
    bias_act_kernel<<<(unsigned)(((long long)N * ENC + threads - 1) / threads), threads, 0, stream>>>(
        gbuf, b2, (long long)N * ENC, ENC, 0);
    aggregate_kernel<ENC><<<aggBlocks, threads, 0, stream>>>(ei, ppmi, degP, XW2p, pbuf, E, N);
    bias_act_kernel<<<(unsigned)(((long long)N * ENC + threads - 1) / threads), threads, 0, stream>>>(
        pbuf, b2, (long long)N * ENC, ENC, 0);

    // 7) fused attention + classifier + domain heads
    float* emb_out = (float*)d_out;                          // [N, 64]
    float* cls_out = emb_out + (size_t)N * ENC;              // [N, 10]
    float* dom_out = cls_out + (size_t)N * NCLS;             // [N, 2]
    head_kernel<<<(N + 127) / 128, 128, 0, stream>>>(
        gbuf, pbuf, att_w, att_b, cls_w, cls_b, dw1, db1, dw2, db2,
        emb_out, cls_out, dom_out, N);
}
